// SAModule_29446295781910
// MI455X (gfx1250) — matrix operand order repaired and checked by reading, not compiled, not run on hardware
//
#include <hip/hip_runtime.h>

typedef __attribute__((ext_vector_type(16))) _Float16 v16h;
typedef __attribute__((ext_vector_type(8)))  _Float16 v8h;
typedef __attribute__((ext_vector_type(4)))  _Float16 v4h;
typedef __attribute__((ext_vector_type(8)))  float    v8f;

#define BATCH 8
#define NPTS  8192
#define DIMF  64
#define SOUT  2048
#define NSAMP 32
#define RAD2  0.16f

#define WMMA_F16(A, B, C) \
  __builtin_amdgcn_wmma_f32_16x16x32_f16(false, (A), false, (B), (short)0, (C), false, false)

// ---------------------------------------------------------------------------
// Phase 1: farthest point sampling. One workgroup per batch; xyz staged in
// LDS (dynamic, 96KB + reduction scratch), running min-distances in regs.
// ---------------------------------------------------------------------------
__global__ __launch_bounds__(256) void fps_kernel(const float* __restrict__ xyz,
                                                  int* __restrict__ fidx) {
  extern __shared__ float smem[];
  float* sx = smem;
  float* sy = sx + NPTS;
  float* sz = sy + NPTS;
  float* rv = sz + NPTS;          // 256 floats
  int*   ri = (int*)(rv + 256);   // 256 ints
  const int b = blockIdx.x, t = threadIdx.x;
  const float* gx = xyz + (size_t)b * 3 * NPTS;
  for (int i = t; i < NPTS; i += 256) {
    sx[i] = gx[i];
    sy[i] = gx[NPTS + i];
    sz[i] = gx[2 * NPTS + i];
  }
  __syncthreads();

  float dist[32];
#pragma unroll
  for (int i = 0; i < 32; ++i) dist[i] = 1e10f;

  int far = 0;
  for (int it = 0; it < SOUT; ++it) {
    if (t == 0) fidx[b * SOUT + it] = far;
    float cx = sx[far], cy = sy[far], cz = sz[far];
    float best = -1.f;
    int bidx = 0;
#pragma unroll
    for (int i = 0; i < 32; ++i) {
      int p = i * 256 + t;
      float dx = sx[p] - cx, dy = sy[p] - cy, dz = sz[p] - cz;
      float d = fminf(dist[i], dx * dx + dy * dy + dz * dz);
      dist[i] = d;
      if (d > best || (d == best && p < bidx)) { best = d; bidx = p; }
    }
    rv[t] = best;
    ri[t] = bidx;
    __syncthreads();
    for (int off = 128; off > 0; off >>= 1) {
      if (t < off) {
        float v2 = rv[t + off];
        int   i2 = ri[t + off];
        if (v2 > rv[t] || (v2 == rv[t] && i2 < ri[t])) { rv[t] = v2; ri[t] = i2; }
      }
      __syncthreads();
    }
    far = ri[0];
    __syncthreads();
  }
}

// ---------------------------------------------------------------------------
// Phase 2: ball query. One wave32 per center; ballot + prefix-popcount gives
// the first NSAMP in-radius indices in ascending order (== sort-then-slice).
// ---------------------------------------------------------------------------
__global__ __launch_bounds__(256) void ballquery_kernel(const float* __restrict__ xyz,
                                                        const int* __restrict__ fidx,
                                                        int* __restrict__ idx,
                                                        float* __restrict__ newxyz) {
  const int wv = threadIdx.x >> 5, lane = threadIdx.x & 31;
  const int c = blockIdx.x * 8 + wv;
  const int b = c / SOUT, s = c % SOUT;
  const float* gx = xyz + (size_t)b * 3 * NPTS;
  const int n0 = fidx[c];
  const float cx = gx[n0], cy = gx[NPTS + n0], cz = gx[2 * NPTS + n0];
  if (lane == 0) {
    newxyz[(b * 3 + 0) * SOUT + s] = cx;
    newxyz[(b * 3 + 1) * SOUT + s] = cy;
    newxyz[(b * 3 + 2) * SOUT + s] = cz;
  }
  int cnt = 0, first = -1;
  for (int nb = 0; nb < NPTS && cnt < NSAMP; nb += 32) {
    int n = nb + lane;
    float dx = gx[n] - cx, dy = gx[NPTS + n] - cy, dz = gx[2 * NPTS + n] - cz;
    bool hit = (dx * dx + dy * dy + dz * dz) <= RAD2;
    unsigned int m = (unsigned int)__ballot(hit);
    if (first < 0 && m) first = nb + __ffs(m) - 1;
    int pos = cnt + __popc(m & ((1u << lane) - 1u));
    if (hit && pos < NSAMP) idx[(size_t)c * NSAMP + pos] = n;
    cnt += __popc(m);
  }
  int total = cnt < NSAMP ? cnt : NSAMP;
  if (lane >= total) idx[(size_t)c * NSAMP + lane] = (first < 0 ? 0 : first);
}

// ---------------------------------------------------------------------------
// WMMA fragment helpers (layouts per CDNA5 ISA 7.12.2, wave32).
// LDS fragment loads are 16-byte ds_load_b128; D results are written as single f16 stores.
// ---------------------------------------------------------------------------
__device__ inline v16h cat8(v8h lo, v8h hi) {
  return __builtin_shufflevector(lo, hi, 0, 1, 2, 3, 4, 5, 6, 7,
                                 8, 9, 10, 11, 12, 13, 14, 15);
}

// A fragment from a row-major f16 tile: lane's halves are rp[0..7], rp[8..15].
__device__ inline v16h load_a(const _Float16* p, int row0, int k0, int ld) {
  const int lane = threadIdx.x & 31;
  const _Float16* rp = p + (row0 + (lane & 15)) * ld + k0 + ((lane >> 4) << 4);
  v8h lo = *(const v8h*)rp;
  v8h hi = *(const v8h*)(rp + 8);
  return cat8(lo, hi);
}

// B fragment from a pre-swizzled [frag][lane][16] buffer: fully contiguous.
__device__ inline v16h load_bfrag(const _Float16* base) {
  const int lane = threadIdx.x & 31;
  const v8h* p = (const v8h*)(base + (lane << 4));
  return cat8(p[0], p[1]);
}

__device__ inline void store_bnrelu(_Float16* y, int row0, int col0, int ld, v8f d,
                                    const float* sc, const float* bi, int cbase) {
  const int lane = threadIdx.x & 31;
  const int col = col0 + (lane & 15);
  const int rb  = row0 + ((lane >> 4) << 3);
  const float s = sc[cbase + col], o = bi[cbase + col];
#pragma unroll
  for (int v = 0; v < 8; ++v) {
    float val = fmaxf(d[v] * s + o, 0.f);
    y[(rb + v) * ld + col] = (_Float16)val;
  }
}

struct BNP {
  const float *g0, *b0, *m0, *v0;
  const float *g1, *b1, *m1, *v1;
  const float *g2, *b2, *m2, *v2;
};

// ---------------------------------------------------------------------------
// Phase 3: gather + 3-layer 1x1-conv MLP (WMMA f16->f32) + maxpool over the
// 32 samples. One wave per center, 2 waves per block; weights staged once per
// block in LDS, pre-swizzled into per-lane B-fragment order.
// Input channels permuted: [64 features][3 rel coords][29 zeros]; W0 rows
// permuted identically, so the math matches the reference concat order.
// ---------------------------------------------------------------------------
__global__ __launch_bounds__(64) void mlp_kernel(const float* __restrict__ xyz,
                                                 const float* __restrict__ pts,
                                                 const float* __restrict__ w0,
                                                 const float* __restrict__ w1,
                                                 const float* __restrict__ w2,
                                                 BNP P,
                                                 const int* __restrict__ idx,
                                                 const float* __restrict__ newxyz,
                                                 float* __restrict__ outp) {
  __shared__ __align__(32) _Float16 W0f[12 * 512];   // frag = nt*3+kt  (KT=3, NT=4)
  __shared__ __align__(32) _Float16 W1f[8 * 512];    // frag = nt*2+kt  (KT=2, NT=4)
  __shared__ __align__(32) _Float16 W2f[16 * 512];   // frag = nt*2+kt  (KT=2, NT=8)
  __shared__ float sc[256], bi[256];
  __shared__ __align__(32) _Float16 Xb[2][32 * 96];
  __shared__ __align__(32) _Float16 Yb[2][32 * 64];

  const int tid = threadIdx.x;
  const int wv = tid >> 5, lane = tid & 31;

  // --- Stage W0 (permuted rows: k<64 -> in-ch 3+k, k in 64..66 -> in-ch k-64).
  for (int i = tid; i < 12 * 512; i += 64) {
    int j = i & 15, ln = (i >> 4) & 31, f = i >> 9;
    int nt = f / 3, kt = f % 3;
    int n = nt * 16 + (ln & 15);
    int k = kt * 32 + ((ln >> 4) << 4) + j;
    float val = 0.f;
    if (k < 64)      val = w0[n * 67 + 3 + k];
    else if (k < 67) val = w0[n * 67 + (k - 64)];
    W0f[i] = (_Float16)val;
  }
  // --- Stage W1.
  for (int i = tid; i < 8 * 512; i += 64) {
    int j = i & 15, ln = (i >> 4) & 31, f = i >> 9;
    int nt = f / 2, kt = f % 2;
    int n = nt * 16 + (ln & 15);
    int k = kt * 32 + ((ln >> 4) << 4) + j;
    W1f[i] = (_Float16)w1[n * 64 + k];
  }
  // --- Stage W2.
  for (int i = tid; i < 16 * 512; i += 64) {
    int j = i & 15, ln = (i >> 4) & 31, f = i >> 9;
    int nt = f / 2, kt = f % 2;
    int n = nt * 16 + (ln & 15);
    int k = kt * 32 + ((ln >> 4) << 4) + j;
    W2f[i] = (_Float16)w2[n * 64 + k];
  }
  // --- Folded BN params.
  for (int c = tid; c < 256; c += 64) {
    const float *g, *bb, *mm, *vv;
    int cc = c;
    if (c < 64)       { g = P.g0; bb = P.b0; mm = P.m0; vv = P.v0; }
    else if (c < 128) { g = P.g1; bb = P.b1; mm = P.m1; vv = P.v1; cc = c - 64; }
    else              { g = P.g2; bb = P.b2; mm = P.m2; vv = P.v2; cc = c - 128; }
    float s = g[cc] * rsqrtf(vv[cc] + 1e-5f);
    sc[c] = s;
    bi[c] = bb[cc] - mm[cc] * s;
  }
  __syncthreads();

  const int c = blockIdx.x * 2 + wv;
  const int b = c >> 11, s = c & (SOUT - 1);
  _Float16* X = Xb[wv];
  _Float16* Y = Yb[wv];

  // --- Gather: lane = sample j; features at cols 0..63, rel coords 64..66.
  {
    const int n = idx[(size_t)c * NSAMP + lane];
    const float cx = newxyz[(b * 3 + 0) * SOUT + s];
    const float cy = newxyz[(b * 3 + 1) * SOUT + s];
    const float cz = newxyz[(b * 3 + 2) * SOUT + s];
    const float* gx = xyz + (size_t)b * 3 * NPTS;
    const float* prow = pts + (size_t)b * DIMF * NPTS + n;
    _Float16* xr = X + lane * 96;
#pragma unroll
    for (int ch = 0; ch < DIMF; ch += 4) {
      v4h t = { (_Float16)prow[(ch + 0) * NPTS], (_Float16)prow[(ch + 1) * NPTS],
                (_Float16)prow[(ch + 2) * NPTS], (_Float16)prow[(ch + 3) * NPTS] };
      *(v4h*)(xr + ch) = t;
    }
    xr[64] = (_Float16)(gx[n] - cx);
    xr[65] = (_Float16)(gx[NPTS + n] - cy);
    xr[66] = (_Float16)(gx[2 * NPTS + n] - cz);
    xr[67] = (_Float16)0.f;
    const v4h z4 = {};
#pragma unroll
    for (int k = 68; k < 96; k += 4) *(v4h*)(xr + k) = z4;
  }
  __syncthreads();

  // --- Layer 1: [32x96] x [96x64] -> Y [32x64], BN+ReLU (channels 0..63).
  for (int mt = 0; mt < 2; ++mt) {
    v16h a0 = load_a(X, mt * 16, 0, 96);
    v16h a1 = load_a(X, mt * 16, 32, 96);
    v16h a2 = load_a(X, mt * 16, 64, 96);
#pragma unroll
    for (int nt = 0; nt < 4; ++nt) {
      v8f acc = {};
      acc = WMMA_F16(a0, load_bfrag(W0f + (nt * 3 + 0) * 512), acc);
      acc = WMMA_F16(a1, load_bfrag(W0f + (nt * 3 + 1) * 512), acc);
      acc = WMMA_F16(a2, load_bfrag(W0f + (nt * 3 + 2) * 512), acc);
      store_bnrelu(Y, mt * 16, nt * 16, 64, acc, sc, bi, 0);
    }
  }
  __syncthreads();

  // --- Layer 2: [32x64] x [64x64] -> X (reused, ld=64), BN+ReLU (64..127).
  for (int mt = 0; mt < 2; ++mt) {
    v16h a0 = load_a(Y, mt * 16, 0, 64);
    v16h a1 = load_a(Y, mt * 16, 32, 64);
#pragma unroll
    for (int nt = 0; nt < 4; ++nt) {
      v8f acc = {};
      acc = WMMA_F16(a0, load_bfrag(W1f + (nt * 2 + 0) * 512), acc);
      acc = WMMA_F16(a1, load_bfrag(W1f + (nt * 2 + 1) * 512), acc);
      store_bnrelu(X, mt * 16, nt * 16, 64, acc, sc, bi, 64);
    }
  }
  __syncthreads();

  // --- Layer 3: [32x64] x [64x128], BN+ReLU (128..255), maxpool over rows.
  {
    v16h x00 = load_a(X, 0, 0, 64),  x01 = load_a(X, 0, 32, 64);
    v16h x10 = load_a(X, 16, 0, 64), x11 = load_a(X, 16, 32, 64);
#pragma unroll
    for (int nt = 0; nt < 8; ++nt) {
      v16h bf0 = load_bfrag(W2f + (nt * 2 + 0) * 512);
      v16h bf1 = load_bfrag(W2f + (nt * 2 + 1) * 512);
      v8f a0 = {}, a1 = {};
      a0 = WMMA_F16(x00, bf0, a0);
      a0 = WMMA_F16(x01, bf1, a0);
      a1 = WMMA_F16(x10, bf0, a1);
      a1 = WMMA_F16(x11, bf1, a1);
      const int col = nt * 16 + (lane & 15);
      const float sv = sc[128 + col], ov = bi[128 + col];
      float mv = 0.f;  // post-ReLU values are >= 0
#pragma unroll
      for (int v = 0; v < 8; ++v) {
        mv = fmaxf(mv, fmaxf(a0[v] * sv + ov, 0.f));
        mv = fmaxf(mv, fmaxf(a1[v] * sv + ov, 0.f));
      }
      mv = fmaxf(mv, __shfl_xor(mv, 16, 32));
      if (lane < 16) outp[((size_t)b * 128 + col) * SOUT + s] = mv;
    }
  }
}

// ---------------------------------------------------------------------------
extern "C" void kernel_launch(void* const* d_in, const int* in_sizes, int n_in,
                              void* d_out, int out_size, void* d_ws, size_t ws_size,
                              hipStream_t stream) {
  (void)in_sizes; (void)n_in; (void)out_size; (void)ws_size;
  const float* xyz    = (const float*)d_in[0];
  const float* points = (const float*)d_in[1];
  const float* w0     = (const float*)d_in[2];
  const float* w1     = (const float*)d_in[3];
  const float* w2     = (const float*)d_in[4];
  BNP P;
  P.g0 = (const float*)d_in[5];  P.b0 = (const float*)d_in[6];
  P.m0 = (const float*)d_in[7];  P.v0 = (const float*)d_in[8];
  P.g1 = (const float*)d_in[9];  P.b1 = (const float*)d_in[10];
  P.m1 = (const float*)d_in[11]; P.v1 = (const float*)d_in[12];
  P.g2 = (const float*)d_in[13]; P.b2 = (const float*)d_in[14];
  P.m2 = (const float*)d_in[15]; P.v2 = (const float*)d_in[16];

  int* fidx = (int*)d_ws;                 // [B*SOUT]
  int* idx  = fidx + BATCH * SOUT;        // [B*SOUT*NSAMP]
  float* out_xyz = (float*)d_out;                       // [B,3,SOUT]
  float* out_np  = out_xyz + BATCH * 3 * SOUT;          // [B,128,SOUT]

  size_t fps_lds = (size_t)(3 * NPTS + 256) * sizeof(float) + 256 * sizeof(int);
  fps_kernel<<<BATCH, 256, fps_lds, stream>>>(xyz, fidx);
  ballquery_kernel<<<(BATCH * SOUT) / 8, 256, 0, stream>>>(xyz, fidx, idx, out_xyz);
  mlp_kernel<<<(BATCH * SOUT) / 2, 64, 0, stream>>>(xyz, points, w0, w1, w2, P,
                                                    idx, out_xyz, out_np);
}
